// VectorQuantizer_3186865733634
// MI455X (gfx1250) — matrix operand order repaired and checked by reading, not compiled, not run on hardware
//
#include <hip/hip_runtime.h>
#include <math.h>

// ---------------------------------------------------------------------------
// VQ-VAE vector quantizer, CDNA5 (gfx1250), wave32 + WMMA f16 16x16x32.
// Distance GEMM uses a hi/lo f16 split (3-term product) for ~f32 accuracy.
// Codebook chunks are staged into LDS by the Tensor Data Mover (TDM),
// double-buffered and overlapped with WMMA compute.
// Sizes hardcoded to the reference: z[32,256,32,32] f32, embedding[1024,256].
// ---------------------------------------------------------------------------

typedef __attribute__((ext_vector_type(16))) _Float16 v16h;
typedef __attribute__((ext_vector_type(8)))  float    v8f;
typedef __attribute__((ext_vector_type(4)))  unsigned int v4u;
typedef __attribute__((ext_vector_type(8)))  int      v8i;
typedef __attribute__((ext_vector_type(4)))  int      v4i;

#define NROWS   32768      // B*H*W
#define DDIM    256
#define KCODE   1024
#define DKA     512        // split-K operand width per row: [hi(256) | lo(256)]
#define LDA     520        // padded LDS row stride (halves): 260 dw == 4 mod 64
#define RPB     64         // rows per block
#define GEMM_THREADS 128   // 4 waves
#define BBYTES  (RPB * LDA * 2u)                      // 66560 per LDS panel
#define GEMM_LDS_BYTES (3u * BBYTES)                  // A + B double buffer

union Frag16 { v16h v; uint4 q[2]; };

// ---- TDM: issue an async 64x512-half (64KB) tile load Global->LDS ----------
// D# group0: count=1 | lds_addr | global_addr(57b) | type=2 (bits 127:126)
// D# group1: data_size=1(2B), pad_enable, pad_interval=7 (256 dw = 1 row),
//            pad_amount=3 (4 dw = 16B)  -> reproduces the LDA=520 row pad.
//            tensor_dim0=512, tensor_dim1=64, tile_dim0=512, tile_dim1=64,
//            tensor_dim0_stride=512 (elements).
__device__ static inline void tdm_load_b(unsigned lds_byte_off, const void* gaddr) {
  unsigned long long ga = (unsigned long long)(uintptr_t)gaddr;
  v4u g0;
  g0[0] = 1u;                                     // count=1 (valid user desc)
  g0[1] = lds_byte_off;                           // LDS destination (bytes)
  g0[2] = (unsigned)ga;                           // global_addr[31:0]
  g0[3] = ((unsigned)(ga >> 32) & 0x01FFFFFFu)    // global_addr[56:32]
        | 0x80000000u;                            // type=2 -> bits 127:126 = 10b
  v8i g1;
  g1[0] = (1 << 16)                               // data_size = 2 bytes
        | (1 << 20)                               // pad_enable
        | (7 << 22)                               // pad_interval: 256 dwords
        | (3 << 25);                              // pad_amount: 4 dwords (16B)
  g1[1] = (int)(512u << 16);                      // tensor_dim0 = 512 (lo16 @ bits63:48)
  g1[2] = (int)(64u << 16);                       // dim0 hi=0 | tensor_dim1 = 64
  g1[3] = (int)(512u << 16);                      // dim1 hi=0 | tile_dim0 = 512
  g1[4] = 64;                                     // tile_dim1 = 64, tile_dim2 = 0
  g1[5] = 512;                                    // tensor_dim0_stride = 512
  g1[6] = 0;
  g1[7] = 0;
  v4i gz = (v4i)0;
#if defined(__clang_major__) && (__clang_major__ >= 23)
  v8i gz8 = (v8i)0;
  __builtin_amdgcn_tensor_load_to_lds(g0, g1, gz, gz, gz8, 0);
#else
  __builtin_amdgcn_tensor_load_to_lds(g0, g1, gz, gz, 0);
#endif
}

// ---------------- init: zero the accumulators we atomically add into --------
__global__ void vq_init(int* __restrict__ counts, float* __restrict__ vecZ,
                        float* __restrict__ sums) {
  int t = threadIdx.x;            // 1024 threads
  counts[t] = 0;
  if (t < DDIM) vecZ[t] = 0.0f;
  if (t < 4)    sums[t] = 0.0f;   // [0]=sum(z^2), [1]=loss_acc
}

// ---------------- per-(b,d) sums of z and z^2 -------------------------------
__global__ void vq_z_sums(const float* __restrict__ z, float* __restrict__ vecZ,
                          float* __restrict__ sums) {
  __shared__ float r1[256], r2[256];
  int t = threadIdx.x;
  const float* p = z + (size_t)blockIdx.x * 1024;   // blockIdx = b*256 + d
  float s1 = 0.0f, s2 = 0.0f;
  for (int i = t; i < 1024; i += 256) { float v = p[i]; s1 += v; s2 += v * v; }
  r1[t] = s1; r2[t] = s2; __syncthreads();
  for (int s = 128; s > 0; s >>= 1) {
    if (t < s) { r1[t] += r1[t + s]; r2[t] += r2[t + s]; }
    __syncthreads();
  }
  if (t == 0) {
    atomicAdd(&vecZ[blockIdx.x & (DDIM - 1)], r1[0]);
    atomicAdd(&sums[0], r2[0]);
  }
}

// ---------------- codebook prep: f16 hi/lo split + ||e_k||^2 ----------------
__global__ void vq_prep_e(const float* __restrict__ emb, _Float16* __restrict__ Bm,
                          float* __restrict__ e_norm) {
  __shared__ float red[256];
  int k = blockIdx.x, d = threadIdx.x;
  float e = emb[k * DDIM + d];
  _Float16 hi = (_Float16)e;
  _Float16 lo = (_Float16)(e - (float)hi);
  Bm[k * DKA + d]        = hi;
  Bm[k * DKA + DDIM + d] = lo;
  red[d] = e * e; __syncthreads();
  for (int s = 128; s > 0; s >>= 1) {
    if (d < s) red[d] += red[d + s];
    __syncthreads();
  }
  if (d == 0) e_norm[k] = red[0];
}

// ---------------- per-d column sums of the codebook -------------------------
__global__ void vq_vec_e(const float* __restrict__ emb, float* __restrict__ vecE) {
  int d = threadIdx.x;  // 256
  float s = 0.0f;
  for (int k = 0; k < KCODE; ++k) s += emb[k * DDIM + d];
  vecE[d] = s;
}

// ---------------- main GEMM + argmin kernel ---------------------------------
extern __shared__ _Float16 gsm[];

__global__ __launch_bounds__(GEMM_THREADS)
void vq_gemm_argmin(const float* __restrict__ z, const _Float16* __restrict__ Bm,
                    const float* __restrict__ e_norm, int* __restrict__ idx_out) {
  _Float16* const Ald   = gsm;                   // 64 x LDA halves (zh | zl)
  _Float16* const Bbuf0 = gsm + RPB * LDA;       // TDM double buffer 0
  _Float16* const Bbuf1 = gsm + 2 * RPB * LDA;   // TDM double buffer 1

  const int tid  = threadIdx.x;
  const int lane = tid & 31;
  const int wave = tid >> 5;
  const int n0   = blockIdx.x * RPB;   // first flat row (b*1024 + hw)
  const int bimg = n0 >> 10;
  const int hw0  = n0 & 1023;

  // Kick off the first TDM codebook chunk before doing any work.
  if (wave == 0) tdm_load_b(BBYTES, Bm);

  // Stage A: on-the-fly transpose + f16 hi/lo split of z rows into LDS
  // (overlaps with the first TDM transfer).
  {
    int r  = tid & 63;
    int d0 = tid >> 6;                 // 0 or 1
    const float* zb = z + ((size_t)bimg * DDIM) * 1024 + hw0 + r;
    for (int d = d0; d < DDIM; d += 2) {
      float v = zb[(size_t)d * 1024];
      _Float16 hi = (_Float16)v;
      _Float16 lo = (_Float16)(v - (float)hi);
      Ald[r * LDA + d]        = hi;
      Ald[r * LDA + DDIM + d] = lo;
    }
  }

  const int lrow  = lane & 15;         // row-in-tile (A) / col-in-tile (B,C)
  const int haf   = lane >> 4;         // which half-wave
  const int aoff  = haf << 3;          // A: +0 / +8 halves inside 32-wide k-tile
  const int boff  = haf << 3;          // B: +0 / +8 halves

  float minv[8];
  int   mini[8];
#pragma unroll
  for (int j = 0; j < 8; ++j) { minv[j] = 3.4e38f; mini[j] = 0; }

  for (int c = 0; c < 16; ++c) {
    if (wave == 0) __builtin_amdgcn_s_wait_tensorcnt((short)0);
    __syncthreads();   // TDM(c) data + A panel visible to everyone
    if ((wave == 0) && (c + 1 < 16))
      tdm_load_b((c & 1) ? BBYTES : 2u * BBYTES,   // fill the *other* buffer
                 Bm + (size_t)(c + 1) * 64 * DKA);

    const _Float16* const Bld = (c & 1) ? Bbuf1 : Bbuf0;
    const int nc = c * 64;

    v8f acc[4];
#pragma unroll
    for (int nt = 0; nt < 4; ++nt) acc[nt] = 0.0f;

    const int arow = (wave << 4) + lrow;
#pragma unroll
    for (int kt = 0; kt < 24; ++kt) {
      // k-tile -> (A segment, B segment) for the 3-term split product
      int ao, bo;
      if (kt < 8)       { ao = kt * 32;                bo = kt * 32; }        // zh*eh
      else if (kt < 16) { ao = DDIM + (kt - 8) * 32;   bo = (kt - 8) * 32; }  // zl*eh
      else              { ao = (kt - 16) * 32;         bo = DDIM + (kt - 16) * 32; } // zh*el

      Frag16 a;
      a.q[0] = *(const uint4*)&Ald[arow * LDA + ao + aoff];       // K 0-7 / 8-15
      a.q[1] = *(const uint4*)&Ald[arow * LDA + ao + 16 + aoff];  // K 16-23 / 24-31
#pragma unroll
      for (int nt = 0; nt < 4; ++nt) {
        Frag16 b;
        const _Float16* bp = &Bld[((nt << 4) + lrow) * LDA + bo + boff];
        b.q[0] = *(const uint4*)bp;
        b.q[1] = *(const uint4*)(bp + 16);
        acc[nt] = __builtin_amdgcn_wmma_f32_16x16x32_f16(
            false, a.v, false, b.v, (short)0, acc[nt], false, false);
      }
    }

    // score = ||e||^2 - 2 z.e ; running argmin (ascending n keeps first-min)
#pragma unroll
    for (int nt = 0; nt < 4; ++nt) {
      int n = nc + (nt << 4) + lrow;
      float en = e_norm[n];
#pragma unroll
      for (int j = 0; j < 8; ++j) {
        float s = en - 2.0f * acc[nt][j];
        if (s < minv[j]) { minv[j] = s; mini[j] = n; }
      }
    }
  }

  // Cross-lane (within each 16-lane half) argmin reduce; tie -> smaller index.
#pragma unroll
  for (int j = 0; j < 8; ++j) {
    float v = minv[j]; int i = mini[j];
#pragma unroll
    for (int m = 1; m <= 8; m <<= 1) {
      float vo = __shfl_xor(v, m, 32);
      int   io = __shfl_xor(i, m, 32);
      if (vo < v || (vo == v && io < i)) { v = vo; i = io; }
    }
    minv[j] = v; mini[j] = i;
  }

  int rbase = n0 + (wave << 4);
  if (lane == 0) {
#pragma unroll
    for (int j = 0; j < 8; ++j) idx_out[rbase + j] = mini[j];
  } else if (lane == 16) {
#pragma unroll
    for (int j = 0; j < 8; ++j) idx_out[rbase + 8 + j] = mini[j];
  }
}

// ---------------- histogram of code usage -----------------------------------
__global__ void vq_hist(const int* __restrict__ idx, int* __restrict__ counts) {
  int n = blockIdx.x * 256 + threadIdx.x;
  atomicAdd(&counts[idx[n]], 1);
}

// ---------------- gather + straight-through output + loss -------------------
__global__ void vq_finalize(const float* __restrict__ z, const float* __restrict__ emb,
                            const int* __restrict__ idx, float* __restrict__ out,
                            float* __restrict__ sums) {
  __shared__ float red[256];
  size_t base = (size_t)blockIdx.x * 8192;
  float ls = 0.0f;
  for (int i = threadIdx.x; i < 8192; i += 256) {
    size_t o = base + i;                      // NCHW flat index
    int bq = (int)(o >> 18);
    int d  = (int)(o >> 10) & 255;
    int hw = (int)o & 1023;
    int n  = (bq << 10) | hw;
    float zp = z[o];
    float zq = emb[idx[n] * DDIM + d];
    float t  = zq - zp;
    out[o] = zp + t;                          // straight-through estimator
    ls += t * t;
  }
  red[threadIdx.x] = ls; __syncthreads();
  for (int s = 128; s > 0; s >>= 1) {
    if ((int)threadIdx.x < s) red[threadIdx.x] += red[threadIdx.x + s];
    __syncthreads();
  }
  if (threadIdx.x == 0) atomicAdd(&sums[1], red[0]);
}

// ---------------- final scalars: loss / perplexity / mean_distance ----------
__global__ void vq_scalars(const int* __restrict__ counts, const float* __restrict__ e_norm,
                           const float* __restrict__ vecZ, const float* __restrict__ vecE,
                           const float* __restrict__ sums, float* __restrict__ out3) {
  __shared__ float red[1024];
  __shared__ float res[3];
  int t = threadIdx.x;

  // entropy term
  float em = (float)counts[t] * (1.0f / 32768.0f);
  red[t] = em * logf(em + 1e-10f);
  __syncthreads();
  for (int s = 512; s > 0; s >>= 1) { if (t < s) red[t] += red[t + s]; __syncthreads(); }
  if (t == 0) res[0] = red[0];
  __syncthreads();

  // sum ||e_k||^2
  red[t] = e_norm[t];
  __syncthreads();
  for (int s = 512; s > 0; s >>= 1) { if (t < s) red[t] += red[t + s]; __syncthreads(); }
  if (t == 0) res[1] = red[0];
  __syncthreads();

  // (sum_n z) . (sum_k e)
  red[t] = (t < DDIM) ? vecZ[t] * vecE[t] : 0.0f;
  __syncthreads();
  for (int s = 512; s > 0; s >>= 1) { if (t < s) red[t] += red[t + s]; __syncthreads(); }
  if (t == 0) res[2] = red[0];
  __syncthreads();

  if (t == 0) {
    float loss = 1.25f * sums[1] / (32768.0f * 256.0f);
    float perp = expf(-res[0]);
    float mean_d = sums[0] / 32768.0f + res[1] / 1024.0f
                 - 2.0f * res[2] / (32768.0f * 1024.0f);
    out3[0] = loss;
    out3[1] = perp;
    out3[2] = mean_d;
  }
}

// ---------------------------------------------------------------------------
extern "C" void kernel_launch(void* const* d_in, const int* in_sizes, int n_in,
                              void* d_out, int out_size, void* d_ws, size_t ws_size,
                              hipStream_t stream) {
  const float* z   = (const float*)d_in[0];   // [32,256,32,32]
  const float* emb = (const float*)d_in[1];   // [1024,256]
  float* out = (float*)d_out;                 // 8388608 + 3

  char* ws = (char*)d_ws;
  _Float16* Bm     = (_Float16*)(ws);                 // 1,048,576 B
  float*    e_norm = (float*)(ws + 1048576);          //     4,096 B
  int*      idx    = (int*)  (ws + 1052672);          //   131,072 B
  int*      counts = (int*)  (ws + 1183744);          //     4,096 B
  float*    vecZ   = (float*)(ws + 1187840);          //     1,024 B
  float*    vecE   = (float*)(ws + 1188864);          //     1,024 B
  float*    sums   = (float*)(ws + 1189888);          //        16 B

  vq_init    <<<1,    1024, 0, stream>>>(counts, vecZ, sums);
  vq_z_sums  <<<8192,  256, 0, stream>>>(z, vecZ, sums);
  vq_prep_e  <<<1024,  256, 0, stream>>>(emb, Bm, e_norm);
  vq_vec_e   <<<1,     256, 0, stream>>>(emb, vecE);
  vq_gemm_argmin<<<NROWS / RPB, GEMM_THREADS, GEMM_LDS_BYTES, stream>>>(z, Bm, e_norm, idx);
  vq_hist    <<<128,   256, 0, stream>>>(idx, counts);
  vq_finalize<<<1024,  256, 0, stream>>>(z, emb, idx, out, sums);
  vq_scalars <<<1,    1024, 0, stream>>>(counts, e_norm, vecZ, vecE, sums,
                                         out + (size_t)8388608);
}
